// myModel_52338471469585
// MI455X (gfx1250) — matrix operand order repaired and checked by reading, not compiled, not run on hardware
//
#include <hip/hip_runtime.h>
#include <math.h>

// ---------------------------------------------------------------------------
// AlphaFold-style gated attention for MI455X (gfx1250, wave32, WMMA).
// All GEMM contractions use v_wmma_f32_16x16x32_bf16 (bf16 operands, f32 acc).
// Per-(b,h) 32-row logits stripe is staged in LDS (~192 KB of the 320 KB/WGP)
// so softmax + weights@V never re-read logits from HBM.
// ---------------------------------------------------------------------------

typedef __attribute__((ext_vector_type(16))) __bf16 v16bf;
typedef __attribute__((ext_vector_type(8)))  float  v8f;
typedef unsigned short u16;
typedef unsigned int   u32;

#define DEV static __device__ __forceinline__

union BF16Frag {
  v16bf v;
  u32   u[8];
};

DEV u16 f2bf(float f) {                      // round-to-nearest-even f32->bf16
  u32 x = __float_as_uint(f);
  x += 0x7FFFu + ((x >> 16) & 1u);
  return (u16)(x >> 16);
}
DEV u32 pack2(float lo, float hi) {
  return (u32)f2bf(lo) | ((u32)f2bf(hi) << 16);
}
DEV v8f wmma_bf16(v16bf a, v16bf b, v8f c) {
  return __builtin_amdgcn_wmma_f32_16x16x32_bf16(false, a, false, b,
                                                 (short)0, c, false, false);
}
DEV v8f v8zero() { v8f z = {0.f,0.f,0.f,0.f,0.f,0.f,0.f,0.f}; return z; }

#define NB   4
#define BQ   1024
#define NH   8
#define HDIM 32
#define QT   32      // q rows per workgroup in the attention kernel

// ---------------------------------------------------------------------------
// Prep: f32 activations -> bf16 (row-major, K contiguous => A-fragments are
// two b128 loads per lane).
// ---------------------------------------------------------------------------
__global__ void k_cvt(const float* __restrict__ q_data,
                      const float* __restrict__ m_data,
                      u32* __restrict__ xq, u32* __restrict__ xm) {
  int i = blockIdx.x * blockDim.x + threadIdx.x;      // one bf16 pair each
  const float* src = blockIdx.y ? m_data : q_data;
  u32* dst = blockIdx.y ? xm : xq;
  float2 f = ((const float2*)src)[i];
  dst[i] = pack2(f.x, f.y);
}

// Prep: (K=256, N=256) f32 weights -> bf16 transposed [N][K] so B-fragments
// are two 16B loads per lane (lane=N, 8 consecutive K each, 16 K apart).
__global__ void k_wprep(const float* __restrict__ w0, const float* __restrict__ w1,
                        const float* __restrict__ w2, const float* __restrict__ w3,
                        const float* __restrict__ w4,
                        u16* __restrict__ t0, u16* __restrict__ t1,
                        u16* __restrict__ t2, u16* __restrict__ t3,
                        u16* __restrict__ t4) {
  int i = blockIdx.x * blockDim.x + threadIdx.x;      // 0..65535
  int k = i >> 8, n = i & 255;
  const float* w; u16* t;
  switch (blockIdx.y) {
    case 0: w = w0; t = t0; break;
    case 1: w = w1; t = t1; break;
    case 2: w = w2; t = t2; break;
    case 3: w = w3; t = t3; break;
    default: w = w4; t = t4; break;
  }
  t[n * 256 + k] = f2bf(w[i]);
}

// ---------------------------------------------------------------------------
// Fused Q/K/V/Gate projections: X(4096x256) @ W(256x256), one 16x16 tile per
// wave, 8 WMMA K-steps. blockIdx.y selects which projection.
//   proj 0: q  -> bf16 [B,H,Q,HD]   (scaled by HD^-0.5)
//   proj 1: k  -> bf16 [B,H,K,HD]
//   proj 2: v  -> bf16 [B,H,HD,K]   (transposed store, b128 per lane)
//   proj 3: gate = sigmoid(. + gating_b) -> f32 [B,Q,256]
// ---------------------------------------------------------------------------
__global__ void k_proj(const u32* __restrict__ xq, const u32* __restrict__ xm,
                       const u16* __restrict__ wqT, const u16* __restrict__ wkT,
                       const u16* __restrict__ wvT, const u16* __restrict__ wgT,
                       const float* __restrict__ gb,
                       u16* __restrict__ qo, u16* __restrict__ ko,
                       u16* __restrict__ vTo, float* __restrict__ gate) {
  const int proj = blockIdx.y;
  const int lane = threadIdx.x & 31;
  const int wv   = threadIdx.x >> 5;
  const int tile = blockIdx.x * 8 + wv;   // 4096 tiles = 256 Mtiles x 16 Ntiles
  const int mt = tile >> 4;
  const int nt = tile & 15;

  const u32* X  = (proj == 1 || proj == 2) ? xm : xq;
  const u16* WT = (proj == 0) ? wqT : (proj == 1) ? wkT : (proj == 2) ? wvT : wgT;

  const int kh   = lane >> 4;                 // K-half selector
  const int mrow = mt * 16 + (lane & 15);     // A row for this lane
  const int ncol = nt * 16 + (lane & 15);     // B col (N) for this lane
  const u32*  arow = X + (size_t)mrow * 128;                  // 128 bf16-pairs/row
  const uint4* brow = (const uint4*)(WT + (size_t)ncol * 256);// 32 uint4/row

  v8f acc = v8zero();
#pragma unroll
  for (int ks = 0; ks < 8; ++ks) {
    BF16Frag A, B;
    // A elems 0-7: K = ks*32 + kh*8 .. +7 ; elems 8-15: +16
    *(uint4*)&A.u[0] = *(const uint4*)(arow + ks * 16 + kh * 4);
    *(uint4*)&A.u[4] = *(const uint4*)(arow + ks * 16 + 8 + kh * 4);
    // B elems 0-7: K = ks*32 + kh*8 .. +7 ; elems 8-15: +16
    *(uint4*)&B.u[0] = brow[ks * 4 + kh];
    *(uint4*)&B.u[4] = brow[ks * 4 + 2 + kh];
    acc = wmma_bf16(A.v, B.v, acc);
  }

  const int n = ncol;
  const int mbase = mt * 16 + kh * 8;   // C rows: M = mbase + r
  const int hh = n >> 5, cc = n & 31;
  if (proj == 0) {
    const float sc = 0.17677669529663687f;  // 32^-0.5
#pragma unroll
    for (int r = 0; r < 8; ++r) {
      int m = mbase + r, bb = m >> 10, qq = m & 1023;
      qo[(((size_t)(bb * NH + hh)) * BQ + qq) * HDIM + cc] = f2bf(acc[r] * sc);
    }
  } else if (proj == 1) {
#pragma unroll
    for (int r = 0; r < 8; ++r) {
      int m = mbase + r, bb = m >> 10, qq = m & 1023;
      ko[(((size_t)(bb * NH + hh)) * BQ + qq) * HDIM + cc] = f2bf(acc[r]);
    }
  } else if (proj == 2) {
    // transposed: vT[b,h,c,k]; lane holds 8 consecutive k for fixed c
    int bb = mbase >> 10, kk = mbase & 1023;
    u32 p[4];
#pragma unroll
    for (int j = 0; j < 4; ++j) p[j] = pack2(acc[2 * j], acc[2 * j + 1]);
    *(uint4*)(vTo + (((size_t)(bb * NH + hh)) * HDIM + cc) * BQ + kk) = *(uint4*)p;
  } else {
    float g = gb[n];
#pragma unroll
    for (int r = 0; r < 8; ++r) {
      int m = mbase + r, bb = m >> 10, qq = m & 1023;
      float s = acc[r] + g;
      gate[((size_t)(bb * BQ + qq)) * 256 + n] = 1.f / (1.f + __expf(-s));
    }
  }
}

// ---------------------------------------------------------------------------
// Attention: one (b, h, 32-row q stripe) per workgroup, 256 threads = 8 waves.
// LDS: 128KB f32 biased logits + 64KB bf16 softmax weights + row sums.
// ---------------------------------------------------------------------------
__global__ void k_attn(const u16* __restrict__ qbf,   // [B,H,Q,32] bf16
                       const u16* __restrict__ kbf,   // [B,H,K,32] bf16
                       const u16* __restrict__ vTbf,  // [B,H,32,K] bf16
                       const float* __restrict__ bias,   // [B,K]
                       const float* __restrict__ nbb,    // [H,Q,K]
                       const float* __restrict__ gate,   // [B,Q,256]
                       float* __restrict__ out_logits,   // [B,H,Q,K] raw
                       u16* __restrict__ gwa) {          // [B,Q,256] bf16
  extern __shared__ char smem[];
  float* Llog = (float*)smem;                              // 32x1024 f32
  u16*   Lwts = (u16*)(smem + QT * BQ * 4);                // 32x1024 bf16
  float* Lsum = (float*)(smem + QT * BQ * 4 + QT * BQ * 2);// [32]
  float* Lred = Llog;                                      // reuse as scratch

  const int tid  = threadIdx.x;
  const int lane = tid & 31;
  const int wv   = tid >> 5;
  const int b = blockIdx.z, h = blockIdx.y;
  const int q0 = blockIdx.x * QT;

  const u16* qmat = qbf + ((size_t)(b * NH + h) * BQ) * HDIM;
  const u16* kmat = kbf + ((size_t)(b * NH + h) * BQ) * HDIM;

  // ---- Phase 1: logits (single WMMA per 16x16 tile; contraction = HD = 32) --
  {
    const int msub = wv & 1;
    const int kh = lane >> 4;
    const int mrow = q0 + msub * 16 + (lane & 15);
    BF16Frag A;
    const uint4* qrow = (const uint4*)(qmat + (size_t)mrow * HDIM); // 64B row
    *(uint4*)&A.u[0] = qrow[kh];
    *(uint4*)&A.u[4] = qrow[2 + kh];

    const int nt0 = wv >> 1;
#pragma unroll 1
    for (int j = 0; j < 16; ++j) {
      const int nt = nt0 + 4 * j;                 // 0..63 k-tiles
      const int kcol = nt * 16 + (lane & 15);
      if (j < 15)  // pull next k-tile toward the caches (global_prefetch_b8)
        __builtin_prefetch(kmat + (size_t)(kcol + 64) * HDIM, 0, 1);
      BF16Frag B;
      const uint4* krow = (const uint4*)(kmat + (size_t)kcol * HDIM);
      *(uint4*)&B.u[0] = krow[kh];
      *(uint4*)&B.u[4] = krow[2 + kh];
      v8f lg = wmma_bf16(A.v, B.v, v8zero());

      const float bc = bias[b * BQ + kcol];
      const int rb = msub * 16 + kh * 8;          // C rows rb..rb+7 in stripe
      float* ol = out_logits + (((size_t)(b * NH + h)) * BQ + (q0 + rb)) * BQ + kcol;
      const float* nb = nbb + ((size_t)h * BQ + (q0 + rb)) * BQ + kcol;
#pragma unroll
      for (int r = 0; r < 8; ++r) {
        float raw = lg[r];
        ol[(size_t)r * BQ] = raw;                             // logits_update
        Llog[(rb + r) * BQ + kcol] = raw + bc + nb[(size_t)r * BQ];
      }
    }
  }
  __syncthreads();

  // ---- Phase 2: softmax. 8 threads/row; unnormalized exp -> bf16 weights. --
  {
    const int row = tid >> 3;
    const int sub = tid & 7;
    const float4* lr = (const float4*)(Llog + row * BQ);
    float m = -3.402823466e38f;
#pragma unroll
    for (int j = 0; j < 32; ++j) {
      float4 v = lr[sub + 8 * j];
      m = fmaxf(m, fmaxf(fmaxf(v.x, v.y), fmaxf(v.z, v.w)));
    }
#pragma unroll
    for (int d = 1; d < 8; d <<= 1) m = fmaxf(m, __shfl_xor(m, d, 32));
    float s = 0.f;
    uint2* wr = (uint2*)(Lwts + row * BQ);
#pragma unroll
    for (int j = 0; j < 32; ++j) {
      float4 v = lr[sub + 8 * j];
      float e0 = __expf(v.x - m), e1 = __expf(v.y - m);
      float e2 = __expf(v.z - m), e3 = __expf(v.w - m);
      s += (e0 + e1) + (e2 + e3);
      uint2 p; p.x = pack2(e0, e1); p.y = pack2(e2, e3);
      wr[sub + 8 * j] = p;                        // 1/sum folded in later
    }
#pragma unroll
    for (int d = 1; d < 8; d <<= 1) s += __shfl_xor(s, d, 32);
    if (sub == 0) Lsum[row] = s;
  }
  __syncthreads();

  // ---- Phase 3: weights(32x1024) @ v(1024x32), split-K across wave pairs. --
  {
    const int t = wv & 3, half = wv >> 2;         // 4 tiles x 2 K-halves
    const int msub = t & 1, ct = t >> 1;
    const int kh = lane >> 4;
    const int c = ct * 16 + (lane & 15);
    const u16* vT   = vTbf + ((size_t)(b * NH + h) * HDIM) * BQ;
    const u16* wrow = Lwts + (msub * 16 + (lane & 15)) * BQ;   // A row in LDS
    const uint4* vrow = (const uint4*)(vT + (size_t)c * BQ);   // 128 uint4/row
    v8f acc = v8zero();
#pragma unroll 1
    for (int i = 0; i < 16; ++i) {
      const int ks = half * 16 + i;
      BF16Frag A, B;
      const uint4* wp = (const uint4*)(wrow + ks * 32);
      *(uint4*)&A.u[0] = wp[kh];
      *(uint4*)&A.u[4] = wp[2 + kh];
      *(uint4*)&B.u[0] = vrow[ks * 4 + kh];
      *(uint4*)&B.u[4] = vrow[ks * 4 + 2 + kh];
      acc = wmma_bf16(A.v, B.v, acc);
    }
    if (half == 1) {                              // spill partial to LDS
      float* dst = Lred + t * 256 + lane * 8;
#pragma unroll
      for (int r = 0; r < 8; ++r) dst[r] = acc[r];
    }
    __syncthreads();
    if (half == 0) {                              // reduce + gate + normalize
      const float* src = Lred + t * 256 + lane * 8;
#pragma unroll
      for (int r = 0; r < 8; ++r) acc[r] += src[r];
#pragma unroll
      for (int r = 0; r < 8; ++r) {
        const int row = msub * 16 + kh * 8 + r;
        const int gq = q0 + row;
        const size_t idx = ((size_t)(b * BQ + gq)) * 256 + h * HDIM + c;
        const float o = acc[r] * (1.f / Lsum[row]) * gate[idx];
        gwa[idx] = f2bf(o);
      }
    }
  }
}

// ---------------------------------------------------------------------------
// Output projection: gwa(4096x256 bf16) @ woT + output_b -> f32 [B,Q,OUT]
// ---------------------------------------------------------------------------
__global__ void k_oproj(const u16* __restrict__ gwa, const u16* __restrict__ woT,
                        const float* __restrict__ ob, float* __restrict__ out) {
  const int lane = threadIdx.x & 31;
  const int wv   = threadIdx.x >> 5;
  const int tile = blockIdx.x * 8 + wv;
  const int mt = tile >> 4, nt = tile & 15;
  const int kh = lane >> 4;
  const int mrow = mt * 16 + (lane & 15);
  const int ncol = nt * 16 + (lane & 15);
  const uint4* arow = (const uint4*)(gwa + (size_t)mrow * 256);
  const uint4* brow = (const uint4*)(woT + (size_t)ncol * 256);
  v8f acc = v8zero();
#pragma unroll
  for (int ks = 0; ks < 8; ++ks) {
    BF16Frag A, B;
    *(uint4*)&A.u[0] = arow[ks * 4 + kh];
    *(uint4*)&A.u[4] = arow[ks * 4 + 2 + kh];
    *(uint4*)&B.u[0] = brow[ks * 4 + kh];
    *(uint4*)&B.u[4] = brow[ks * 4 + 2 + kh];
    acc = wmma_bf16(A.v, B.v, acc);
  }
  const float bo = ob[ncol];
  const int mbase = mt * 16 + kh * 8;
#pragma unroll
  for (int r = 0; r < 8; ++r)
    out[(size_t)(mbase + r) * 256 + ncol] = acc[r] + bo;
}

// ---------------------------------------------------------------------------
extern "C" void kernel_launch(void* const* d_in, const int* in_sizes, int n_in,
                              void* d_out, int out_size, void* d_ws, size_t ws_size,
                              hipStream_t stream) {
  const float* q_data   = (const float*)d_in[0];
  const float* m_data   = (const float*)d_in[1];
  const float* bias     = (const float*)d_in[2];
  const float* nbb      = (const float*)d_in[3];
  const float* query_w  = (const float*)d_in[4];
  const float* key_w    = (const float*)d_in[5];
  const float* value_w  = (const float*)d_in[6];
  const float* gating_w = (const float*)d_in[7];
  const float* gating_b = (const float*)d_in[8];
  const float* output_w = (const float*)d_in[9];
  const float* output_b = (const float*)d_in[10];
  (void)in_sizes; (void)n_in; (void)out_size; (void)ws_size;

  // Workspace carve-up (~16.7 MB total), 256B aligned slices.
  char* ws = (char*)d_ws;
  size_t off = 0;
  auto take = [&](size_t bytes) {
    char* p = ws + off;
    off += (bytes + 255) & ~(size_t)255;
    return p;
  };
  u32* xq   = (u32*)take((size_t)4096 * 128 * 4);   // q_data bf16 pairs
  u32* xm   = (u32*)take((size_t)4096 * 128 * 4);   // m_data bf16 pairs
  u16* wqT  = (u16*)take((size_t)65536 * 2);
  u16* wkT  = (u16*)take((size_t)65536 * 2);
  u16* wvT  = (u16*)take((size_t)65536 * 2);
  u16* wgT  = (u16*)take((size_t)65536 * 2);
  u16* woT  = (u16*)take((size_t)65536 * 2);
  u16* qbf  = (u16*)take((size_t)NB * NH * BQ * HDIM * 2);
  u16* kbf  = (u16*)take((size_t)NB * NH * BQ * HDIM * 2);
  u16* vTbf = (u16*)take((size_t)NB * NH * BQ * HDIM * 2);
  float* gate = (float*)take((size_t)NB * BQ * 256 * 4);
  u16* gwa  = (u16*)take((size_t)NB * BQ * 256 * 2);

  float* out        = (float*)d_out;                 // [B,Q,OUT]
  float* out_logits = out + (size_t)NB * BQ * 256;   // [B,H,Q,K]

  k_cvt<<<dim3(2048, 2), 256, 0, stream>>>(q_data, m_data, xq, xm);
  k_wprep<<<dim3(256, 5), 256, 0, stream>>>(query_w, key_w, value_w, gating_w,
                                            output_w, wqT, wkT, wvT, wgT, woT);
  k_proj<<<dim3(512, 4), 256, 0, stream>>>(xq, xm, wqT, wkT, wvT, wgT, gating_b,
                                           qbf, kbf, vTbf, gate);

  const int LDSB = QT * BQ * 4 + QT * BQ * 2 + 256;  // 196864 B (< 320 KB/WGP)
  hipFuncSetAttribute(reinterpret_cast<const void*>(&k_attn),
                      hipFuncAttributeMaxDynamicSharedMemorySize, LDSB);
  k_attn<<<dim3(BQ / QT, NH, NB), 256, LDSB, stream>>>(
      qbf, kbf, vTbf, bias, nbb, gate, out_logits, gwa);

  k_oproj<<<512, 256, 0, stream>>>(gwa, woT, output_b, out);
}
